// TemplatePairStack_91345364451463
// MI455X (gfx1250) — compile-verified, matrix operand order checked by reading, not run on hardware
//
#include <hip/hip_runtime.h>
#include <hip/hip_bf16.h>
#include <math.h>

typedef __bf16 bf16;
typedef __attribute__((ext_vector_type(16))) __bf16 v16bf;
typedef __attribute__((ext_vector_type(8)))  __bf16 v8bf;
typedef __attribute__((ext_vector_type(2)))  __bf16 v2bf;
typedef __attribute__((ext_vector_type(8)))  float  v8f;

#define NRES 256
#define CH   64
#define NH   4
#define HD   16
#define NROWS (NRES*NRES)   // 65536
#define FF   256            // transition hidden = TF*C

// ---------------------------------------------------------------------------
// WMMA helpers (CDNA5 gfx1250, wave32). D = A(16x32) * B(32x16) + C, bf16->f32
// ---------------------------------------------------------------------------
__device__ inline v8f wmma_bf16(v16bf a, v16bf b, v8f c) {
  return __builtin_amdgcn_wmma_f32_16x16x32_bf16(
      /*neg_a=*/false, a, /*neg_b=*/false, b,
      /*c_mod=*/(short)0, c, /*reuse_a=*/false, /*reuse_b=*/false);
}

__device__ inline v16bf cat8(v8bf lo, v8bf hi) {
  v16bf a;
#pragma unroll
  for (int i = 0; i < 8; ++i) { a[i] = lo[i]; a[8+i] = hi[i]; }
  return a;
}

// A tile 16(M) x 32(K), K-contiguous: A[m][k] = p[m*mStride + k].
// ISA 7.12.2 layout: lanes 0-15 hold K 0..7 & 16..23; lanes 16-31 hold K 8..15 & 24..31.
// Per lane: two 16B vector loads. p and mStride must be 8-element (16B) aligned.
__device__ inline v16bf load_a_k1(const bf16* p, int mStride) {
  int l = threadIdx.x & 31;
  const bf16* q = p + (l & 15)*mStride + ((l >> 4) << 3);
  return cat8(*(const v8bf*)q, *(const v8bf*)(q + 16));
}

// B tile 32(K) x 16(N), K-contiguous: B[k][n] = p[n*nStride + k]. Same per-lane shape.
__device__ inline v16bf load_b_k1(const bf16* p, int nStride) {
  int l = threadIdx.x & 31;
  const bf16* q = p + (l & 15)*nStride + ((l >> 4) << 3);
  return cat8(*(const v8bf*)q, *(const v8bf*)(q + 16));
}

__device__ inline float sigmoidf_(float x) { return 1.f / (1.f + __expf(-x)); }

// ---------------------------------------------------------------------------
// Utility kernels
// ---------------------------------------------------------------------------
__global__ void k_copy(const float* __restrict__ s, float* __restrict__ d, int n) {
  int i = blockIdx.x*blockDim.x + threadIdx.x;
  if (i < n) d[i] = s[i];
}

// Transposing f32 -> bf16 weight convert: src is [K][N] row-major, dst is [N][K].
__global__ void k_convert_t(const float* __restrict__ s, bf16* __restrict__ d,
                            int K, int N) {
  int i = blockIdx.x*blockDim.x + threadIdx.x;
  if (i < K*N) { int k = i / N, n = i - k*N; d[n*K + k] = (bf16)s[i]; }
}

// feat2d_w [64][4] -> transposed, zero-padded bf16 [16][64]
__global__ void k_pad_feat2d_t(const float* __restrict__ s, bf16* __restrict__ d) {
  int i = blockIdx.x*blockDim.x + threadIdx.x;
  if (i < 16*64) {
    int n = i >> 6, k = i & 63;
    d[i] = (bf16)((n < NH) ? s[k*NH + n] : 0.f);
  }
}

// LayerNorm over C=64 -> bf16, optional transpose-read and mask pre-multiply.
__global__ void k_ln_bf16(const float* __restrict__ src, const float* __restrict__ scale,
                          const float* __restrict__ offset, const float* __restrict__ mask,
                          bf16* __restrict__ dst, int transposed, int usemask) {
  int wave = threadIdx.x >> 5, lane = threadIdx.x & 31;
  int m = blockIdx.x*8 + wave;
  int b = m >> 8, q = m & 255;
  int rpos = transposed ? ((q << 8) | b) : m;
  float mk = usemask ? mask[rpos] : 1.f;
  int c0 = lane * 2;
  float2 xv = *(const float2*)(src + rpos*CH + c0);
  float x0 = xv.x * mk, x1 = xv.y * mk;
  float s = x0 + x1, s2 = x0*x0 + x1*x1;
#pragma unroll
  for (int o = 16; o >= 1; o >>= 1) {
    s  += __shfl_xor(s,  o, 32);
    s2 += __shfl_xor(s2, o, 32);
  }
  float mu  = s * (1.f/CH);
  float var = s2 * (1.f/CH) - mu*mu;
  float rs  = rsqrtf(var + 1e-5f);
  v2bf o2;
  o2[0] = (bf16)((x0 - mu)*rs*scale[c0]   + offset[c0]);
  o2[1] = (bf16)((x1 - mu)*rs*scale[c0+1] + offset[c0+1]);
  *(v2bf*)(dst + m*CH + c0) = o2;
}

// ---------------------------------------------------------------------------
// Attention kernels
// ---------------------------------------------------------------------------
// Q/K zero-padded to K=32 per head: [B][H][256][32]; V transposed: [B][H][16][256];
// gate logits G f32 [65536][64]. One wave per (mtile, which); A loaded once, 4 N-tiles.
__global__ void k_qkvg(const bf16* __restrict__ actn,
                       const bf16* __restrict__ wqT, const bf16* __restrict__ wkT,
                       const bf16* __restrict__ wvT, const bf16* __restrict__ wgT,
                       bf16* __restrict__ Qp, bf16* __restrict__ Kp,
                       bf16* __restrict__ VmT, float* __restrict__ G) {
  int gw = blockIdx.x*8 + (threadIdx.x >> 5);
  int mt = gw >> 2, which = gw & 3;   // 0=q 1=k 2=v 3=gate
  const bf16* wT = (which==0) ? wqT : (which==1) ? wkT : (which==2) ? wvT : wgT;
  const bf16* ab = actn + mt*16*CH;
  v16bf a0 = load_a_k1(ab, CH), a1 = load_a_k1(ab + 32, CH);
  int l = threadIdx.x & 31, n = l & 15, m0 = (l>>4)<<3;
  float scl = (which == 0) ? 0.25f : 1.f;   // D^-0.5
#pragma unroll
  for (int nt = 0; nt < 4; ++nt) {
    v8f c = {};
    c = wmma_bf16(a0, load_b_k1(wT + nt*16*CH,      CH), c);
    c = wmma_bf16(a1, load_b_k1(wT + nt*16*CH + 32, CH), c);
#pragma unroll
    for (int j = 0; j < 8; ++j) {
      int gr = mt*16 + m0 + j;
      int b = gr >> 8, qq = gr & 255;
      float v = c[j] * scl;
      if (which == 0) {
        bf16* p = Qp + (((b*NH + nt)*NRES + qq)*32 + n);
        p[0] = (bf16)v; p[16] = (bf16)0.f;
      } else if (which == 1) {
        bf16* p = Kp + (((b*NH + nt)*NRES + qq)*32 + n);
        p[0] = (bf16)v; p[16] = (bf16)0.f;
      } else if (which == 2) {
        VmT[((b*NH + nt)*HD + n)*NRES + qq] = (bf16)v;
      } else {
        G[gr*CH + nt*16 + n] = v;
      }
    }
  }
}

// pair bias: nbt[q*256+k][h] = sum_c actn[q,k,c]*feat2d[c,h]
__global__ void k_nb(const bf16* __restrict__ actn, const bf16* __restrict__ f2pT,
                     float* __restrict__ nbt) {
  int gw = blockIdx.x*8 + (threadIdx.x >> 5);
  const bf16* ab = actn + gw*16*CH;
  v8f c = {};
  c = wmma_bf16(load_a_k1(ab,      CH), load_b_k1(f2pT,      CH), c);
  c = wmma_bf16(load_a_k1(ab + 32, CH), load_b_k1(f2pT + 32, CH), c);
  int l = threadIdx.x & 31, n = l & 15, m0 = (l>>4)<<3;
  if (n < NH)
#pragma unroll
    for (int j = 0; j < 8; ++j) nbt[(gw*16 + m0 + j)*NH + n] = c[j];
}

// Fused logits + softmax + AV + gate. One block (8 waves) per (qtile, h, b).
__global__ void __launch_bounds__(256)
k_attn(const bf16* __restrict__ Qp, const bf16* __restrict__ Kp,
       const bf16* __restrict__ VmT, const float* __restrict__ G,
       const float* __restrict__ nbt, const float* __restrict__ mask,
       bf16* __restrict__ waG, int transposed) {
  __shared__ __align__(16) bf16 P[16*256];
  __shared__ float smax[8][16], ssum[8][16], gmax[16], gsum[16];
  __shared__ float pacc[8][256];
  int wave = threadIdx.x >> 5, lane = threadIdx.x & 31;
  int q0 = blockIdx.x * 16, h = blockIdx.y, b = blockIdx.z;
  int n = lane & 15, m0 = (lane >> 4) << 3;

  const bf16* qb = Qp + ((b*NH + h)*NRES + q0)*32;
  v16bf A = load_a_k1(qb, 32);
  int ct0 = wave * 2;
  v8f c0 = {}, c1 = {};
  c0 = wmma_bf16(A, load_b_k1(Kp + ((b*NH+h)*NRES + ct0*16)*32,     32), c0);
  c1 = wmma_bf16(A, load_b_k1(Kp + ((b*NH+h)*NRES + (ct0+1)*16)*32, 32), c1);

  float pm[8];
#pragma unroll
  for (int j = 0; j < 8; ++j) {
    int qg = q0 + m0 + j;
    int k0c = ct0*16 + n, k1c = k0c + 16;
    float mk0 = transposed ? mask[k0c*NRES + b] : mask[b*NRES + k0c];
    float mk1 = transposed ? mask[k1c*NRES + b] : mask[b*NRES + k1c];
    c0[j] += 1e9f*(mk0 - 1.f) + nbt[(qg*NRES + k0c)*NH + h];
    c1[j] += 1e9f*(mk1 - 1.f) + nbt[(qg*NRES + k1c)*NH + h];
    pm[j]  = fmaxf(c0[j], c1[j]);
  }
#pragma unroll
  for (int o = 8; o >= 1; o >>= 1)
#pragma unroll
    for (int j = 0; j < 8; ++j) pm[j] = fmaxf(pm[j], __shfl_xor(pm[j], o, 32));
  if (n == 0)
#pragma unroll
    for (int j = 0; j < 8; ++j) smax[wave][m0 + j] = pm[j];
  __syncthreads();
  if (threadIdx.x < 16) {
    float mx = smax[0][threadIdx.x];
    for (int w = 1; w < 8; ++w) mx = fmaxf(mx, smax[w][threadIdx.x]);
    gmax[threadIdx.x] = mx;
  }
  __syncthreads();

  float ps[8];
#pragma unroll
  for (int j = 0; j < 8; ++j) {
    float mx = gmax[m0 + j];
    c0[j] = __expf(c0[j] - mx);
    c1[j] = __expf(c1[j] - mx);
    ps[j] = c0[j] + c1[j];
  }
#pragma unroll
  for (int o = 8; o >= 1; o >>= 1)
#pragma unroll
    for (int j = 0; j < 8; ++j) ps[j] += __shfl_xor(ps[j], o, 32);
  if (n == 0)
#pragma unroll
    for (int j = 0; j < 8; ++j) ssum[wave][m0 + j] = ps[j];
  __syncthreads();
  if (threadIdx.x < 16) {
    float s = 0.f;
    for (int w = 0; w < 8; ++w) s += ssum[w][threadIdx.x];
    gsum[threadIdx.x] = s;
  }
  __syncthreads();
#pragma unroll
  for (int j = 0; j < 8; ++j) {
    float inv = 1.f / gsum[m0 + j];
    P[(m0 + j)*256 + ct0*16 + n]     = (bf16)(c0[j] * inv);
    P[(m0 + j)*256 + (ct0+1)*16 + n] = (bf16)(c1[j] * inv);
  }
  __syncthreads();

  // AV: each wave one 32-key chunk; deterministic LDS tree-sum.
  int kc = wave * 32;
  v8f cav = {};
  cav = wmma_bf16(load_a_k1(P + kc, 256),
                  load_b_k1(VmT + (b*NH + h)*HD*NRES + kc, NRES), cav);
#pragma unroll
  for (int j = 0; j < 8; ++j) pacc[wave][(m0 + j)*16 + n] = cav[j];
  __syncthreads();

  int r = threadIdx.x >> 4, d = threadIdx.x & 15;
  float wa = 0.f;
#pragma unroll
  for (int w = 0; w < 8; ++w) wa += pacc[w][r*16 + d];
  int gi = (b*NRES + q0 + r)*CH + h*16 + d;
  waG[gi] = (bf16)(wa * sigmoidf_(G[gi]));
}

// act[pos] += waG @ out_w  (un-transposes for per-column attention)
__global__ void k_outproj(const bf16* __restrict__ waG, const bf16* __restrict__ owT,
                          float* __restrict__ act, int transposed) {
  int mt = blockIdx.x*8 + (threadIdx.x >> 5);
  const bf16* ab = waG + mt*16*CH;
  v16bf a0 = load_a_k1(ab, CH), a1 = load_a_k1(ab + 32, CH);
  int l = threadIdx.x & 31, n = l & 15, m0 = (l>>4)<<3;
#pragma unroll
  for (int nt = 0; nt < 4; ++nt) {
    v8f c = {};
    c = wmma_bf16(a0, load_b_k1(owT + nt*16*CH,      CH), c);
    c = wmma_bf16(a1, load_b_k1(owT + nt*16*CH + 32, CH), c);
#pragma unroll
    for (int j = 0; j < 8; ++j) {
      int gr = mt*16 + m0 + j;
      int b = gr >> 8, qq = gr & 255;
      int pos = transposed ? ((qq << 8) | b) : gr;
      act[pos*CH + nt*16 + n] += c[j];
    }
  }
}

// ---------------------------------------------------------------------------
// Triangle multiplication kernels
// ---------------------------------------------------------------------------
// left/right = mask * (a@W + b) * sigmoid(a@Wg + bg), written channel-major
// cm[c][x][y] and transposed cmt[c][y][x] so the triangle einsum is K-contiguous.
__global__ void k_leftright(const bf16* __restrict__ a, const bf16* __restrict__ wT,
                            const float* __restrict__ bias, const bf16* __restrict__ gwT,
                            const float* __restrict__ gbias, const float* __restrict__ mask,
                            bf16* __restrict__ cm, bf16* __restrict__ cmt) {
  int mt = blockIdx.x*8 + (threadIdx.x >> 5);
  const bf16* ab = a + mt*16*CH;
  v16bf a0 = load_a_k1(ab, CH), a1 = load_a_k1(ab + 32, CH);
  int l = threadIdx.x & 31, n = l & 15, m0 = (l>>4)<<3;
#pragma unroll
  for (int nt = 0; nt < 4; ++nt) {
    v8f c = {}, g = {};
    c = wmma_bf16(a0, load_b_k1(wT  + nt*16*CH,      CH), c);
    c = wmma_bf16(a1, load_b_k1(wT  + nt*16*CH + 32, CH), c);
    g = wmma_bf16(a0, load_b_k1(gwT + nt*16*CH,      CH), g);
    g = wmma_bf16(a1, load_b_k1(gwT + nt*16*CH + 32, CH), g);
    int col = nt*16 + n;
#pragma unroll
    for (int j = 0; j < 8; ++j) {
      int gr = mt*16 + m0 + j;
      bf16 bv = (bf16)(mask[gr] * (c[j] + bias[col]) * sigmoidf_(g[j] + gbias[col]));
      cm [col*NROWS + gr] = bv;
      cmt[col*NROWS + ((gr & 255) << 8) + (gr >> 8)] = bv;
    }
  }
}

// X[i,j,c] = sum_k A[i,k]*B[j,k] per channel. A = Acm[c][m][k], B = Bcm[c][n][k],
// both K-contiguous. outgoing: (left_cm, right_cm); incoming: (right_cmT, left_cmT).
// One wave per (itile, channel): A row of 8 tiles cached in regs, 16 jt x 8 WMMA.
__global__ void k_triangle(const bf16* __restrict__ Acm, const bf16* __restrict__ Bcm,
                           float* __restrict__ X) {
  int gw = blockIdx.x*8 + (threadIdx.x >> 5);   // 0..1023
  int c = gw & 63, it = gw >> 6;
  const bf16* Ab = Acm + c*NROWS + it*16*NRES;
  v16bf A8[8];
#pragma unroll
  for (int s = 0; s < 8; ++s) A8[s] = load_a_k1(Ab + s*32, NRES);
  int l = threadIdx.x & 31, n = l & 15, m0 = (l>>4)<<3;
  for (int jt = 0; jt < 16; ++jt) {
    const bf16* Bb = Bcm + c*NROWS + jt*16*NRES;
    if (jt < 15) __builtin_prefetch(Bb + 16*NRES, 0, 1);
    v8f acc = {};
#pragma unroll
    for (int s = 0; s < 8; ++s)
      acc = wmma_bf16(A8[s], load_b_k1(Bb + s*32, NRES), acc);
#pragma unroll
    for (int j = 0; j < 8; ++j)
      X[((it*16 + m0 + j)*NRES + jt*16 + n)*CH + c] = acc[j];
  }
}

// act += (xn @ out_w + out_b) * sigmoid(a @ gate_w + gate_b)
__global__ void k_tmfinal(const bf16* __restrict__ xn, const bf16* __restrict__ owT,
                          const float* __restrict__ ob, const bf16* __restrict__ a,
                          const bf16* __restrict__ gwT, const float* __restrict__ gb,
                          float* __restrict__ act) {
  int mt = blockIdx.x*8 + (threadIdx.x >> 5);
  const bf16* xb = xn + mt*16*CH;
  const bf16* ab = a  + mt*16*CH;
  v16bf x0 = load_a_k1(xb, CH), x1 = load_a_k1(xb + 32, CH);
  v16bf a0 = load_a_k1(ab, CH), a1 = load_a_k1(ab + 32, CH);
  int l = threadIdx.x & 31, n = l & 15, m0 = (l>>4)<<3;
#pragma unroll
  for (int nt = 0; nt < 4; ++nt) {
    v8f c = {}, g = {};
    c = wmma_bf16(x0, load_b_k1(owT + nt*16*CH,      CH), c);
    c = wmma_bf16(x1, load_b_k1(owT + nt*16*CH + 32, CH), c);
    g = wmma_bf16(a0, load_b_k1(gwT + nt*16*CH,      CH), g);
    g = wmma_bf16(a1, load_b_k1(gwT + nt*16*CH + 32, CH), g);
    int col = nt*16 + n;
#pragma unroll
    for (int j = 0; j < 8; ++j) {
      int gr = mt*16 + m0 + j;
      act[gr*CH + col] += (c[j] + ob[col]) * sigmoidf_(g[j] + gb[col]);
    }
  }
}

// ---------------------------------------------------------------------------
// Transition kernels
// ---------------------------------------------------------------------------
// h = swish(t@w1) * (t@w2)  -> bf16 [65536][256]; w1T/w2T are [256][64]
__global__ void k_trhidden(const bf16* __restrict__ t, const bf16* __restrict__ w1T,
                           const bf16* __restrict__ w2T, bf16* __restrict__ hbuf) {
  int mt = blockIdx.x*8 + (threadIdx.x >> 5);
  const bf16* ab = t + mt*16*CH;
  v16bf a0 = load_a_k1(ab, CH), a1 = load_a_k1(ab + 32, CH);
  int l = threadIdx.x & 31, n = l & 15, m0 = (l>>4)<<3;
  for (int nt = 0; nt < 16; ++nt) {
    v8f ca = {}, cb = {};
    ca = wmma_bf16(a0, load_b_k1(w1T + nt*16*CH,      CH), ca);
    ca = wmma_bf16(a1, load_b_k1(w1T + nt*16*CH + 32, CH), ca);
    cb = wmma_bf16(a0, load_b_k1(w2T + nt*16*CH,      CH), cb);
    cb = wmma_bf16(a1, load_b_k1(w2T + nt*16*CH + 32, CH), cb);
#pragma unroll
    for (int j = 0; j < 8; ++j) {
      int gr = mt*16 + m0 + j;
      float av = ca[j];
      hbuf[gr*FF + nt*16 + n] = (bf16)(av * sigmoidf_(av) * cb[j]);
    }
  }
}

// act += h @ w3 ; w3T is [64][256]
__global__ void k_trfinal(const bf16* __restrict__ hbuf, const bf16* __restrict__ w3T,
                          float* __restrict__ act) {
  int mt = blockIdx.x*8 + (threadIdx.x >> 5);
  const bf16* ab = hbuf + mt*16*FF;
  v16bf A8[8];
#pragma unroll
  for (int s = 0; s < 8; ++s) A8[s] = load_a_k1(ab + s*32, FF);
  int l = threadIdx.x & 31, n = l & 15, m0 = (l>>4)<<3;
#pragma unroll
  for (int nt = 0; nt < 4; ++nt) {
    v8f c = {};
#pragma unroll
    for (int s = 0; s < 8; ++s)
      c = wmma_bf16(A8[s], load_b_k1(w3T + nt*16*FF + s*32, FF), c);
#pragma unroll
    for (int j = 0; j < 8; ++j) {
      int gr = mt*16 + m0 + j;
      act[gr*CH + nt*16 + n] += c[j];
    }
  }
}

// ---------------------------------------------------------------------------
// Host orchestration
// ---------------------------------------------------------------------------
#define MB (size_t(1) << 20)
static const size_t OFF_ACT  = 0;         // f32 act             16 MB
static const size_t OFF_ACTN = 16*MB;     // bf16 actn/a/t        8 MB
static const size_t OFF_B1   = 24*MB;     // Qpad | left_cm | xn | hidden(32MB -> 56MB)
static const size_t OFF_B1B  = 32*MB;     // right_cm
static const size_t OFF_B2   = 40*MB;     // Kpad | x f32        16 MB
static const size_t OFF_B3   = 56*MB;     // VmT | left_cmT       8 MB
static const size_t OFF_B3B  = 64*MB;     // waG | right_cmT      8 MB
static const size_t OFF_G    = 72*MB;     // gate logits f32     16 MB
static const size_t OFF_NBT  = 88*MB;     // pair bias f32        1 MB
static const size_t OFF_W    = 89*MB;     // bf16 weights        <1 MB

extern "C" void kernel_launch(void* const* d_in, const int* in_sizes, int n_in,
                              void* d_out, int out_size, void* d_ws, size_t ws_size,
                              hipStream_t stream) {
  (void)in_sizes; (void)n_in; (void)out_size; (void)ws_size;
  char* ws = (char*)d_ws;
  float* act   = (float*)(ws + OFF_ACT);
  bf16*  actn  = (bf16*)(ws + OFF_ACTN);
  bf16*  Qp    = (bf16*)(ws + OFF_B1);
  bf16*  Kp    = (bf16*)(ws + OFF_B2);
  bf16*  VmT   = (bf16*)(ws + OFF_B3);
  bf16*  waG   = (bf16*)(ws + OFF_B3B);
  float* G     = (float*)(ws + OFF_G);
  float* nbt   = (float*)(ws + OFF_NBT);
  bf16*  wbuf  = (bf16*)(ws + OFF_W);
  bf16*  lcm   = (bf16*)(ws + OFF_B1);
  bf16*  rcm   = (bf16*)(ws + OFF_B1B);
  bf16*  lcmt  = (bf16*)(ws + OFF_B3);
  bf16*  rcmt  = (bf16*)(ws + OFF_B3B);
  float* X     = (float*)(ws + OFF_B2);
  bf16*  xn    = (bf16*)(ws + OFF_B1);   // after k_triangle, left_cm is dead
  bf16*  hbuf  = (bf16*)(ws + OFF_B1);   // 32MB spans B1..B2

  const float* pair_act = (const float*)d_in[0];
  const float* mask     = (const float*)d_in[1];

  k_copy<<<NROWS*CH/256, 256, 0, stream>>>(pair_act, act, NROWS*CH);

  for (int blk = 0; blk < 2; ++blk) {
    int base = 2 + blk*53;  // tas(8) tae(8) tmo(16) tmi(16) tr(5)

    // ---- triangle attention: t=0 starting (per-row), t=1 ending (per-column)
    for (int t = 0; t < 2; ++t) {
      int pb = base + t*8;
      const float* ln_s = (const float*)d_in[pb+0];
      const float* ln_o = (const float*)d_in[pb+1];
      const float* f2   = (const float*)d_in[pb+2];
      const float* qw   = (const float*)d_in[pb+3];
      const float* kw   = (const float*)d_in[pb+4];
      const float* vw   = (const float*)d_in[pb+5];
      const float* gwp  = (const float*)d_in[pb+6];
      const float* ow   = (const float*)d_in[pb+7];
      int transposed = t;

      k_ln_bf16<<<8192, 256, 0, stream>>>(act, ln_s, ln_o, nullptr, actn, transposed, 0);
      bf16 *wqT = wbuf, *wkT = wbuf+4096, *wvT = wbuf+8192,
           *wgT = wbuf+12288, *owT = wbuf+16384, *f2pT = wbuf+20480;
      k_convert_t<<<16, 256, 0, stream>>>(qw,  wqT, CH, CH);
      k_convert_t<<<16, 256, 0, stream>>>(kw,  wkT, CH, CH);
      k_convert_t<<<16, 256, 0, stream>>>(vw,  wvT, CH, CH);
      k_convert_t<<<16, 256, 0, stream>>>(gwp, wgT, CH, CH);
      k_convert_t<<<16, 256, 0, stream>>>(ow,  owT, CH, CH);
      k_pad_feat2d_t<<<4, 256, 0, stream>>>(f2, f2pT);

      k_qkvg<<<2048, 256, 0, stream>>>(actn, wqT, wkT, wvT, wgT, Qp, Kp, VmT, G);
      k_nb<<<512, 256, 0, stream>>>(actn, f2pT, nbt);
      dim3 ag(16, NH, NRES);
      k_attn<<<ag, 256, 0, stream>>>(Qp, Kp, VmT, G, nbt, mask, waG, transposed);
      k_outproj<<<512, 256, 0, stream>>>(waG, owT, act, transposed);
    }

    // ---- triangle multiplication: t=0 outgoing (tmo), t=1 incoming (tmi)
    for (int t = 0; t < 2; ++t) {
      int pb = base + 16 + t*16;
      int outgoing = (t == 0);
      const float* lns = (const float*)d_in[pb+0];
      const float* lno = (const float*)d_in[pb+1];
      const float* lw  = (const float*)d_in[pb+2];
      const float* lb  = (const float*)d_in[pb+3];
      const float* rw  = (const float*)d_in[pb+4];
      const float* rb  = (const float*)d_in[pb+5];
      const float* lgw = (const float*)d_in[pb+6];
      const float* lgb = (const float*)d_in[pb+7];
      const float* rgw = (const float*)d_in[pb+8];
      const float* rgb = (const float*)d_in[pb+9];
      const float* cs  = (const float*)d_in[pb+10];
      const float* co  = (const float*)d_in[pb+11];
      const float* ow  = (const float*)d_in[pb+12];
      const float* ob  = (const float*)d_in[pb+13];
      const float* gwp = (const float*)d_in[pb+14];
      const float* gb  = (const float*)d_in[pb+15];

      k_ln_bf16<<<8192, 256, 0, stream>>>(act, lns, lno, nullptr, actn, 0, 0);
      bf16 *wlT = wbuf, *wlgT = wbuf+4096, *wrT = wbuf+8192,
           *wrgT = wbuf+12288, *woT = wbuf+16384, *wgT = wbuf+20480;
      k_convert_t<<<16, 256, 0, stream>>>(lw,  wlT,  CH, CH);
      k_convert_t<<<16, 256, 0, stream>>>(lgw, wlgT, CH, CH);
      k_convert_t<<<16, 256, 0, stream>>>(rw,  wrT,  CH, CH);
      k_convert_t<<<16, 256, 0, stream>>>(rgw, wrgT, CH, CH);
      k_convert_t<<<16, 256, 0, stream>>>(ow,  woT,  CH, CH);
      k_convert_t<<<16, 256, 0, stream>>>(gwp, wgT,  CH, CH);

      k_leftright<<<512, 256, 0, stream>>>(actn, wlT, lb, wlgT, lgb, mask, lcm, lcmt);
      k_leftright<<<512, 256, 0, stream>>>(actn, wrT, rb, wrgT, rgb, mask, rcm, rcmt);
      if (outgoing) k_triangle<<<128, 256, 0, stream>>>(lcm,  rcm,  X);
      else          k_triangle<<<128, 256, 0, stream>>>(rcmt, lcmt, X);
      k_ln_bf16<<<8192, 256, 0, stream>>>(X, cs, co, nullptr, xn, 0, 0);
      k_tmfinal<<<512, 256, 0, stream>>>(xn, woT, ob, actn, wgT, gb, act);
    }

    // ---- transition
    {
      int pb = base + 48;
      const float* lns = (const float*)d_in[pb+0];
      const float* lno = (const float*)d_in[pb+1];
      const float* w1  = (const float*)d_in[pb+2];
      const float* w2  = (const float*)d_in[pb+3];
      const float* w3  = (const float*)d_in[pb+4];

      k_ln_bf16<<<8192, 256, 0, stream>>>(act, lns, lno, mask, actn, 0, 1);
      bf16 *w1T = wbuf, *w2T = wbuf+16384, *w3T = wbuf+32768;
      k_convert_t<<<64, 256, 0, stream>>>(w1, w1T, CH, FF);   // -> [256][64]
      k_convert_t<<<64, 256, 0, stream>>>(w2, w2T, CH, FF);   // -> [256][64]
      k_convert_t<<<64, 256, 0, stream>>>(w3, w3T, FF, CH);   // -> [64][256]
      k_trhidden<<<512, 256, 0, stream>>>(actn, w1T, w2T, hbuf);
      k_trfinal<<<512, 256, 0, stream>>>(hbuf, w3T, act);
    }
  }

  k_copy<<<NROWS*CH/256, 256, 0, stream>>>(act, (float*)d_out, NROWS*CH);
}
